// MultiScanVSSM_84327387890656
// MI455X (gfx1250) — matrix operand order repaired and checked by reading, not compiled, not run on hardware
//
#include <hip/hip_runtime.h>
#include <hip/hip_bf16.h>
#include <math.h>

typedef __attribute__((ext_vector_type(16))) _Float16 v16h;
typedef __attribute__((ext_vector_type(8)))  float    v8f;
typedef unsigned int u32;
typedef __attribute__((ext_vector_type(4))) u32 v4u;
typedef __attribute__((ext_vector_type(8))) int v8i;
typedef __attribute__((ext_vector_type(4))) int v4i;

#define Bsz  32
#define Cch  384
#define Ll   1024
#define EPSF 1e-8f

__device__ inline v8f wmma16x16x32(v16h a, v16h b, v8f c) {
  return __builtin_amdgcn_wmma_f32_16x16x32_f16(false, a, false, b, (short)0, c, false, false);
}
__device__ inline float gelu_exact(float v) { return 0.5f * v * (1.0f + erff(v * 0.70710678118654752f)); }
__device__ inline float sigm(float v) { return 1.0f / (1.0f + expf(-v)); }
// y = x + alpha * clip(wts*(x+EPS), -5, 5)
__device__ inline float yrecomb(float xv, float wt, float alpha) {
  float r = wt * (xv + EPSF);
  r = fminf(fmaxf(r, -5.0f), 5.0f);
  return xv + alpha * r;
}

// ---------------------------------------------------------------------------
// K1: forward 2D-DFT magnitudes via WMMA; per-image low/high band energies.
// One wave32 per 32x32 image; 8 images per 256-thread block.
// Image is DMA'd into LDS by the Tensor Data Mover (tensor_load_to_lds),
// then converted f32 -> f16 for the WMMA DFT pipeline.
// ---------------------------------------------------------------------------
__global__ __launch_bounds__(256) void freq_energy_kernel(const float* __restrict__ x,
                                                          float* __restrict__ energy) {
  __shared__ _Float16 Cm[32][32];
  __shared__ _Float16 Sm[32][32];
  __shared__ _Float16 Xs[8][32][32];
  // per-wave 4KB scratch: first holds the TDM-loaded f32 image, later reused
  // (after the image is consumed) as the two f16 intermediate tiles T1/T2.
  __shared__ __align__(128) unsigned char scratch[8][4096];

  const int tid = threadIdx.x;
  for (int i = tid; i < 1024; i += 256) {
    int k = i >> 5, n = i & 31;
    float ang = (float)((k * n) & 31) * 0.19634954084936207f;  // 2*pi/32
    Cm[k][n] = (_Float16)(cosf(ang) * 0.17677669529663687f);   // 1/sqrt(32) ortho
    Sm[k][n] = (_Float16)(sinf(ang) * 0.17677669529663687f);
  }

  const int wv   = tid >> 5;
  const int lane = tid & 31;
  const int img  = blockIdx.x * 8 + wv;

  float* Xf = (float*)&scratch[wv][0];
  _Float16 (*T1)[32] = (_Float16(*)[32])&scratch[wv][0];
  _Float16 (*T2)[32] = (_Float16(*)[32])&scratch[wv][2048];

  // ---- Tensor Data Mover: DMA this wave's 32x32 f32 tile into LDS ----
  {
    u32 lds_off = (u32)(uintptr_t)(void*)Xf;  // flat addr[31:0] == logical LDS address
    unsigned long long ga = (unsigned long long)(uintptr_t)(x + (size_t)img * 1024);
    u32 glo = (u32)ga, ghi = (u32)(ga >> 32);
    lds_off = __builtin_amdgcn_readfirstlane((int)lds_off);
    glo     = __builtin_amdgcn_readfirstlane((int)glo);
    ghi     = __builtin_amdgcn_readfirstlane((int)ghi);

    v4u g0;
    g0[0] = 1u;                                   // count=1, user descriptor
    g0[1] = lds_off;                              // lds_addr [63:32]
    g0[2] = glo;                                  // global_addr [95:64]
    g0[3] = (ghi & 0x01FFFFFFu) | (2u << 30);     // global_addr[56:32] | type=2

    v8i g1;
    g1[0] = 0x00020000;   // workgroup_mask=0, data_size=2 (4B)
    g1[1] = (32 << 16);   // tensor_dim0 = 32 (bits 79:48, low half)
    g1[2] = (32 << 16);   // tensor_dim0 hi=0 | tensor_dim1 = 32 (low half)
    g1[3] = (32 << 16);   // tensor_dim1 hi=0 | tile_dim0 = 32
    g1[4] = 32;           // tile_dim1 = 32, tile_dim2 = 0
    g1[5] = 32;           // tensor_dim0_stride = 32 (low 32 of 48)
    g1[6] = 0;            // stride0 hi | tensor_dim1_stride lo
    g1[7] = 0;

    v4i gz  = {0, 0, 0, 0};          // groups 2/3 unused (<=2D tensor)
    v8i gz8 = {0, 0, 0, 0, 0, 0, 0, 0};
    __builtin_amdgcn_tensor_load_to_lds(g0, g1, gz, gz, gz8, 0);
  }
  __builtin_amdgcn_s_wait_tensorcnt(0);

  // convert the DMA'd f32 tile to f16 (+EPS) for WMMA consumption
#pragma unroll
  for (int i = 0; i < 32; ++i)
    Xs[wv][i][lane] = (_Float16)(Xf[i * 32 + lane] + EPSF);
  __syncthreads();

  const int m   = lane & 15;
  const int kh2 = lane >> 4;

  // A fragments (16-bit A 16x32 layout): K = kh2*8 + (e<8 ? e : e+8)
  v16h aC[2], aS[2], bX[2];
#pragma unroll
  for (int i = 0; i < 2; ++i) {
#pragma unroll
    for (int e = 0; e < 16; ++e) {
      int K = kh2 * 8 + ((e < 8) ? e : (e + 8));
      aC[i][e] = Cm[16 * i + m][K];
      aS[i][e] = Sm[16 * i + m][K];
    }
  }
  // B fragments (N on lanes, K on elements)
#pragma unroll
  for (int j = 0; j < 2; ++j) {
#pragma unroll
    for (int e = 0; e < 16; ++e) bX[j][e] = Xs[wv][kh2 * 8 + ((e < 8) ? e : (e + 8))][16 * j + m];
  }

  v8f zero = {};
  // Stage 1: T1 = C*X, T2 = S*X  (overwrites the consumed f32 tile)
#pragma unroll
  for (int i = 0; i < 2; ++i) {
#pragma unroll
    for (int j = 0; j < 2; ++j) {
      v8f t1 = wmma16x16x32(aC[i], bX[j], zero);
      v8f t2 = wmma16x16x32(aS[i], bX[j], zero);
#pragma unroll
      for (int r = 0; r < 8; ++r) {
        int row = 16 * i + r + 8 * kh2;
        int col = 16 * j + m;
        T1[row][col] = (_Float16)t1[r];
        T2[row][col] = (_Float16)t2[r];
      }
    }
  }
  __syncthreads();

  // Stage 2 fragments (negations folded in; f16 WMMA has no A/B NEG bits)
  v16h aT1[2], aT1n[2], aT2n[2], bC[2], bS[2];
#pragma unroll
  for (int i = 0; i < 2; ++i) {
#pragma unroll
    for (int e = 0; e < 16; ++e) {
      int K = kh2 * 8 + ((e < 8) ? e : (e + 8));
      _Float16 v1 = T1[16 * i + m][K];
      _Float16 v2 = T2[16 * i + m][K];
      aT1[i][e]  = v1;
      aT1n[i][e] = -v1;
      aT2n[i][e] = -v2;
    }
  }
#pragma unroll
  for (int j = 0; j < 2; ++j) {
#pragma unroll
    for (int e = 0; e < 16; ++e) {
      bC[j][e] = Cm[kh2 * 8 + ((e < 8) ? e : (e + 8))][16 * j + m];  // C, S symmetric => C^T == C
      bS[j][e] = Sm[kh2 * 8 + ((e < 8) ? e : (e + 8))][16 * j + m];
    }
  }

  float lowsum = 0.f, highsum = 0.f;
#pragma unroll
  for (int i = 0; i < 2; ++i) {
#pragma unroll
    for (int j = 0; j < 2; ++j) {
      v8f re = wmma16x16x32(aT2n[i], bS[j], zero);  // Re = T1*C - T2*S
      re = wmma16x16x32(aT1[i], bC[j], re);
      v8f im = wmma16x16x32(aT2n[i], bC[j], zero);  // Im = -(T1*S + T2*C)
      im = wmma16x16x32(aT1n[i], bS[j], im);
#pragma unroll
      for (int r = 0; r < 8; ++r) {
        int kh = 16 * i + r + 8 * kh2;
        int kw = 16 * j + m;
        if (kw < 17) {
          float mg = sqrtf(re[r] * re[r] + im[r] * im[r]) + EPSF;
          if (kh < 8 && kw < 4) lowsum += mg;            // mag[:8, :4]
          if (kh >= 8 && kw >= 4) highsum += mg;         // mag[8:, 4:17]
        }
      }
    }
  }
#pragma unroll
  for (int off = 16; off > 0; off >>= 1) {
    lowsum  += __shfl_xor(lowsum, off, 32);
    highsum += __shfl_xor(highsum, off, 32);
  }
  if (lane == 0) {
    int b = img / Cch, c = img % Cch;
    energy[(size_t)b * (2 * Cch) + c]       = lowsum  * (1.0f / 32.0f);
    energy[(size_t)b * (2 * Cch) + Cch + c] = highsum * (1.0f / 312.0f);
  }
}

// ---------------------------------------------------------------------------
// K2: freq MLP  wts = sigmoid(gelu(energy @ w1 + b1) @ w2 + b2)   (768->24->384)
// ---------------------------------------------------------------------------
__global__ __launch_bounds__(256) void freq_mlp_kernel(const float* __restrict__ energy,
                                                       const float* __restrict__ w1,
                                                       const float* __restrict__ b1,
                                                       const float* __restrict__ w2,
                                                       const float* __restrict__ b2,
                                                       float* __restrict__ wts) {
  const int b = blockIdx.x, t = threadIdx.x;
  __shared__ float e[768];
  __shared__ float h[24];
  for (int i = t; i < 768; i += 256) e[i] = energy[(size_t)b * 768 + i];
  __syncthreads();
  if (t < 24) {
    float a = b1[t];
    for (int k = 0; k < 768; ++k) a += e[k] * w1[k * 24 + t];
    h[t] = gelu_exact(a);
  }
  __syncthreads();
  for (int c = t; c < Cch; c += 256) {
    float a = b2[c];
#pragma unroll
    for (int j = 0; j < 24; ++j) a += h[j] * w2[j * Cch + c];
    wts[(size_t)b * Cch + c] = sigm(a);
  }
}

// ---------------------------------------------------------------------------
// K3: per-(b,l) LayerNorm stats over C, then xg[b,c] = mean_l(xn) * g + beta
// ---------------------------------------------------------------------------
__global__ __launch_bounds__(1024) void stats_kernel(const float* __restrict__ x,
                                                     const float* __restrict__ wts,
                                                     const float* __restrict__ ln_g,
                                                     const float* __restrict__ ln_b,
                                                     const float* __restrict__ alpha_p,
                                                     float* __restrict__ xg) {
  __shared__ float muS[Ll];
  __shared__ float rsS[Ll];
  const int b = blockIdx.x, t = threadIdx.x;
  const float alpha = alpha_p[0];
  const float* xb = x + (size_t)b * Cch * Ll;

  // phase 1: thread t handles spatial position l = t (coalesced across threads)
  float s = 0.f, s2 = 0.f;
  for (int c = 0; c < Cch; ++c) {
    float v = yrecomb(xb[(size_t)c * Ll + t], wts[(size_t)b * Cch + c], alpha);
    s += v; s2 += v * v;
  }
  float mu = s * (1.0f / (float)Cch);
  float var = s2 * (1.0f / (float)Cch) - mu * mu;
  muS[t] = mu;
  rsS[t] = rsqrtf(var + 1e-5f);
  __syncthreads();

  // phase 2: thread t = channel c, stream its contiguous row
  if (t < Cch) {
    const float wt = wts[(size_t)b * Cch + t];
    const float* row = xb + (size_t)t * Ll;
    float acc = 0.f;
    for (int l = 0; l < Ll; ++l) {
      __builtin_prefetch(row + l + 64, 0, 1);
      float v = yrecomb(row[l], wt, alpha);
      acc += (v - muS[l]) * rsS[l];
    }
    xg[(size_t)b * Cch + t] = acc * (1.0f / (float)Ll) * ln_g[t] + ln_b[t];
  }
}

// ---------------------------------------------------------------------------
// K4: attn MLP  attn = sigmoid(gelu(xg @ wg + bg) @ wc + bc)   (384->48->384)
// ---------------------------------------------------------------------------
__global__ __launch_bounds__(256) void attn_mlp_kernel(const float* __restrict__ xg,
                                                       const float* __restrict__ wg,
                                                       const float* __restrict__ bg,
                                                       const float* __restrict__ wc,
                                                       const float* __restrict__ bc,
                                                       float* __restrict__ attn) {
  const int b = blockIdx.x, t = threadIdx.x;
  __shared__ float e[Cch];
  __shared__ float h[48];
  for (int i = t; i < Cch; i += 256) e[i] = xg[(size_t)b * Cch + i];
  __syncthreads();
  if (t < 48) {
    float a = bg[t];
    for (int k = 0; k < Cch; ++k) a += e[k] * wg[k * 48 + t];
    h[t] = gelu_exact(a);
  }
  __syncthreads();
  for (int c = t; c < Cch; c += 256) {
    float a = bc[c];
#pragma unroll
    for (int j = 0; j < 48; ++j) a += h[j] * wc[j * Cch + c];
    attn[(size_t)b * Cch + c] = sigm(a);
  }
}

// ---------------------------------------------------------------------------
// K5: out[b,l,c] = 4 * y[b,c,l] * attn[b,c]  (LDS-tiled transpose)
// grid (L/32, C/32, B), block (32, 8)
// ---------------------------------------------------------------------------
__global__ __launch_bounds__(256) void out_kernel(const float* __restrict__ x,
                                                  const float* __restrict__ wts,
                                                  const float* __restrict__ attn,
                                                  const float* __restrict__ alpha_p,
                                                  float* __restrict__ out) {
  __shared__ float tile[32][33];
  const int b = blockIdx.z;
  const int l0 = blockIdx.x * 32;
  const int c0 = blockIdx.y * 32;
  const float alpha = alpha_p[0];
  const float* xb = x + ((size_t)b * Cch + c0) * Ll + l0;
#pragma unroll
  for (int it = 0; it < 4; ++it) {
    int ci = threadIdx.y + 8 * it;
    float wt = wts[(size_t)b * Cch + c0 + ci];
    tile[ci][threadIdx.x] = yrecomb(xb[(size_t)ci * Ll + threadIdx.x], wt, alpha);
  }
  __syncthreads();
  const float at = 4.0f * attn[(size_t)b * Cch + c0 + threadIdx.x];
#pragma unroll
  for (int it = 0; it < 4; ++it) {
    int li = threadIdx.y + 8 * it;
    out[((size_t)b * Ll + l0 + li) * Cch + c0 + threadIdx.x] = tile[threadIdx.x][li] * at;
  }
}

// ---------------------------------------------------------------------------
extern "C" void kernel_launch(void* const* d_in, const int* in_sizes, int n_in,
                              void* d_out, int out_size, void* d_ws, size_t ws_size,
                              hipStream_t stream) {
  const float* x       = (const float*)d_in[0];
  const float* freq_w1 = (const float*)d_in[1];
  const float* freq_b1 = (const float*)d_in[2];
  const float* freq_w2 = (const float*)d_in[3];
  const float* freq_b2 = (const float*)d_in[4];
  const float* alpha   = (const float*)d_in[5];
  const float* ln_g    = (const float*)d_in[6];
  const float* ln_b    = (const float*)d_in[7];
  const float* wg      = (const float*)d_in[8];
  const float* bg      = (const float*)d_in[9];
  const float* wc      = (const float*)d_in[10];
  const float* bc      = (const float*)d_in[11];
  float* out = (float*)d_out;

  float* wsf    = (float*)d_ws;
  float* energy = wsf;                    // 32*768
  float* wts    = wsf + 32 * 768;         // 32*384
  float* xg     = wts + 32 * 384;         // 32*384
  float* attn   = xg + 32 * 384;          // 32*384

  freq_energy_kernel<<<(Bsz * Cch) / 8, 256, 0, stream>>>(x, energy);
  freq_mlp_kernel<<<Bsz, 256, 0, stream>>>(energy, freq_w1, freq_b1, freq_w2, freq_b2, wts);
  stats_kernel<<<Bsz, 1024, 0, stream>>>(x, wts, ln_g, ln_b, alpha, xg);
  attn_mlp_kernel<<<Bsz, 256, 0, stream>>>(xg, wg, bg, wc, bc, attn);
  out_kernel<<<dim3(Ll / 32, Cch / 32, Bsz), dim3(32, 8), 0, stream>>>(x, wts, attn, alpha, out);
}
